// ContrastiveAtt_Block_54279796687156
// MI455X (gfx1250) — matrix operand order repaired and checked by reading, not compiled, not run on hardware
//
#include <hip/hip_runtime.h>
#include <hip/hip_fp16.h>
#include <math.h>

typedef __attribute__((ext_vector_type(16))) _Float16 v16h;
typedef __attribute__((ext_vector_type(8)))  _Float16 v8h;
typedef __attribute__((ext_vector_type(8)))  float    v8f;
typedef __attribute__((ext_vector_type(4)))  float    v4f;

namespace {

constexpr int    Bn    = 16;
constexpr int    Cch   = 128;
constexpr int    HWn   = 4096;
constexpr int    DHn   = 256;     // HEADS*ATTD
constexpr int    QKVn  = 512;
constexpr int    HIDn  = 340;
constexpr int    FFN2  = 680;     // 2*HID
constexpr float  SCALEq = 0.25f;  // KD^-0.5

// -------------------------------------------------------------------------
// Depthwise 3x3 'SAME' conv, optional per-channel scale g / shift b,
// optional residual (out = x + conv + b), optional relu.
// -------------------------------------------------------------------------
__global__ __launch_bounds__(256)
void dwconv3x3_kernel(const float* __restrict__ x,
                      const float* __restrict__ w,   // (C,1,3,3)
                      const float* __restrict__ g,   // (C,) or null
                      const float* __restrict__ b,   // (C,) or null
                      float* __restrict__ y,
                      int Cn, int residual, int relu, int total)
{
  int idx = blockIdx.x * 256 + threadIdx.x;
  if (idx >= total) return;
  int wv = idx & 63;
  int hv = (idx >> 6) & 63;
  int c  = (idx >> 12) % Cn;
  int bt = idx / (Cn << 12);
  const float* xp = x + ((size_t)bt * Cn + c) * HWn;
  const float* wc = w + c * 9;
  float acc = 0.0f;
  if (hv >= 1 && hv < 63 && wv >= 1 && wv < 63) {
    const float* ctr = xp + hv * 64 + wv;
#pragma unroll
    for (int dy = -1; dy <= 1; ++dy)
#pragma unroll
      for (int dx = -1; dx <= 1; ++dx)
        acc += wc[(dy + 1) * 3 + (dx + 1)] * ctr[dy * 64 + dx];
  } else {
#pragma unroll
    for (int dy = -1; dy <= 1; ++dy) {
#pragma unroll
      for (int dx = -1; dx <= 1; ++dx) {
        int hh = hv + dy, ww = wv + dx;
        if (hh >= 0 && hh < 64 && ww >= 0 && ww < 64)
          acc += wc[(dy + 1) * 3 + (dx + 1)] * xp[hh * 64 + ww];
      }
    }
  }
  if (g) acc *= g[c];
  if (b) acc += b[c];
  if (residual) acc += xp[hv * 64 + wv];
  if (relu) acc = fmaxf(acc, 0.0f);
  y[idx] = acc;
}

// -------------------------------------------------------------------------
// LayerNorm over the channel dim of NCHW (C=128 fixed).
// -------------------------------------------------------------------------
__global__ __launch_bounds__(256)
void ln_c_kernel(const float* __restrict__ x,
                 const float* __restrict__ g,
                 const float* __restrict__ b,
                 float* __restrict__ y)
{
  int pix = blockIdx.x * 256 + threadIdx.x;       // 0..B*HW-1
  if (pix >= Bn * HWn) return;
  int bt = pix / HWn;
  int hw = pix - bt * HWn;
  const float* xp = x + (size_t)bt * Cch * HWn + hw;
  float m = 0.0f;
  for (int c = 0; c < Cch; ++c) m += xp[(size_t)c * HWn];
  m *= (1.0f / Cch);
  float v = 0.0f;
  for (int c = 0; c < Cch; ++c) { float d = xp[(size_t)c * HWn] - m; v += d * d; }
  v *= (1.0f / Cch);
  float r = rsqrtf(v + 1e-6f);
  float* yp = y + (size_t)bt * Cch * HWn + hw;
  for (int c = 0; c < Cch; ++c)
    yp[(size_t)c * HWn] = (xp[(size_t)c * HWn] - m) * r * g[c] + b[c];
}

// -------------------------------------------------------------------------
// WMMA f16 GEMM for 1x1 conv + folded BN + activation epilogue.
//   Y[bat, c_off+m, n] = act( g[m]*(sum_k W[m,k]*maybe_relu(X[bat,k,n])) + b[m] )
//                       (+ addsrc at same index if given)
// Block: 256 threads = 8 waves. Block tile 64(M) x 64(N), K-step 32.
// Waves 2(M) x 4(N); each wave: 2 A fragments + 1 B fragment -> 2 v_wmma/step.
// Double-buffered LDS (one barrier per K-step); next tile's global loads are
// issued before the current tile's WMMAs so HBM latency hides behind math.
// W staged [m][k], X staged transposed [n][k]: every fragment is two
// contiguous 16B ds_load_b128 (CDNA5 16-bit A/B VGPR layouts).
// -------------------------------------------------------------------------
__global__ __launch_bounds__(256)
void wmma_gemm_kernel(const float* __restrict__ Wm,   // (O,K) row-major
                      const float* __restrict__ X,    // (B,K,HWc)
                      const float* __restrict__ g,    // (O,) or null
                      const float* __restrict__ bb,   // (O,) or null
                      const float* __restrict__ addsrc,
                      float* __restrict__ Y,          // (B,CoutTot,HWc)
                      int O, int K, int HWc,
                      int CoutTot, int c_off,
                      int relu_in, int act_out)
{
  __shared__ __align__(16) _Float16 Wt[2][64][40];   // [buf][m][k], row 80B
  __shared__ __align__(16) _Float16 Xt[2][64][40];   // [buf][n][k] transposed

  const int bat  = blockIdx.z;
  const int n0   = blockIdx.x * 64;
  const int m0   = blockIdx.y * 64;
  const int tid  = threadIdx.x;
  const int lane = tid & 31;
  const int wave = tid >> 5;        // 0..7 (wave32)
  const int wm   = wave >> 2;       // 0..1 -> 32-row M slab
  const int wn   = wave & 3;        // 0..3 -> 16-col N slab

  const float* Xb = X + (size_t)bat * K * HWc;

  // staging coordinates: thread -> (row, 8 contiguous k)
  const int srow = tid & 63;        // m for Wt, n for Xt
  const int skq  = tid >> 6;        // 0..3 -> k = skq*8 .. +7

  // fragment coordinates (hoisted)
  const int klo = (lane < 16) ? 0 : 8;      // A: K halves 0..7 | 8..15
  const int khi = (lane < 16) ? 16 : 24;    // A: K halves 16..23 | 24..31
  const int mr0 = wm * 32 + (lane & 15);
  const int nr  = wn * 16 + (lane & 15);
  const int kb  = (lane < 16) ? 0 : 16;     // (unused) B uses klo/khi like A

  v8f acc0 = {};
  v8f acc1 = {};

  // staging registers (next tile)
  v4f rw0, rw1;
  float rx[8];

  auto ld = [&](int k0) {           // issue global loads for tile k0
    const float* wrow = Wm + (size_t)(m0 + srow) * K + k0 + skq * 8;
    rw0 = *(const v4f*)wrow;
    rw1 = *(const v4f*)(wrow + 4);
    const float* xcol = Xb + (size_t)(k0 + skq * 8) * HWc + n0 + srow;
#pragma unroll
    for (int j = 0; j < 8; ++j) rx[j] = xcol[(size_t)j * HWc];
    __builtin_prefetch(xcol + 32 * (size_t)HWc, 0, 1);   // global_prefetch_b8
  };
  auto st = [&](int buf) {          // convert + store staged tile to LDS
    v8h hw_, hx;
#pragma unroll
    for (int j = 0; j < 4; ++j) { hw_[j] = (_Float16)rw0[j]; hw_[4 + j] = (_Float16)rw1[j]; }
#pragma unroll
    for (int j = 0; j < 8; ++j) {
      float v = rx[j];
      if (relu_in) v = fmaxf(v, 0.0f);
      hx[j] = (_Float16)v;
    }
    *(v8h*)&Wt[buf][srow][skq * 8] = hw_;
    *(v8h*)&Xt[buf][srow][skq * 8] = hx;
  };
  auto compute = [&](int buf) {     // 2 x v_wmma on tile in LDS buf
    v8h a0lo = *(const v8h*)&Wt[buf][mr0][klo];
    v8h a0hi = *(const v8h*)&Wt[buf][mr0][khi];
    v8h a1lo = *(const v8h*)&Wt[buf][mr0 + 16][klo];
    v8h a1hi = *(const v8h*)&Wt[buf][mr0 + 16][khi];
    v16h a0 = __builtin_shufflevector(a0lo, a0hi, 0,1,2,3,4,5,6,7,8,9,10,11,12,13,14,15);
    v16h a1 = __builtin_shufflevector(a1lo, a1hi, 0,1,2,3,4,5,6,7,8,9,10,11,12,13,14,15);
    v8h b0 = *(const v8h*)&Xt[buf][nr][klo];
    v8h b1 = *(const v8h*)&Xt[buf][nr][khi];
    v16h bf = __builtin_shufflevector(b0, b1, 0,1,2,3,4,5,6,7,8,9,10,11,12,13,14,15);
    acc0 = __builtin_amdgcn_wmma_f32_16x16x32_f16(false, a0, false, bf, (short)0, acc0, false, false);
    acc1 = __builtin_amdgcn_wmma_f32_16x16x32_f16(false, a1, false, bf, (short)0, acc1, false, false);
  };

  const bool mfull = (m0 + 64 <= O);
  const int  Kmain = mfull ? (K & ~31) : 0;   // branch-free main region

  if (Kmain > 0) {
    ld(0);
    st(0);
    __syncthreads();
    int cur = 0;
    for (int k0 = 32; k0 < Kmain; k0 += 32) {
      ld(k0);            // global loads issue first
      compute(cur);      // WMMAs hide load latency
      st(cur ^ 1);       // waits loads, converts, stores other buffer
      __syncthreads();   // single barrier per K-step
      cur ^= 1;
    }
    compute(cur);
  }

  // tail: K remainder and/or M-edge blocks (clamped loads + select-zero)
  if (Kmain < K) {
    const int mg  = m0 + srow;
    const int mgc = (mg < O) ? mg : (O - 1);
    for (int k0 = Kmain; k0 < K; k0 += 32) {
      __syncthreads();
#pragma unroll
      for (int j = 0; j < 8; ++j) {
        int kg  = k0 + skq * 8 + j;
        int kgc = (kg < K) ? kg : (K - 1);
        float wv = Wm[(size_t)mgc * K + kgc];
        if (!(kg < K && mg < O)) wv = 0.0f;
        Wt[0][srow][skq * 8 + j] = (_Float16)wv;
        float xv = Xb[(size_t)kgc * HWc + n0 + srow];
        if (relu_in) xv = fmaxf(xv, 0.0f);
        if (!(kg < K)) xv = 0.0f;
        Xt[0][srow][skq * 8 + j] = (_Float16)xv;
      }
      __syncthreads();
      compute(0);
    }
  }

  // D layout: VGPR r -> M = r (+8 for lanes 16-31), N = lane&15.
  const int nl    = (lane & 15);
  const int mbase = (lane < 16) ? 0 : 8;
#pragma unroll
  for (int half = 0; half < 2; ++half) {
    v8f acc = half ? acc1 : acc0;
#pragma unroll
    for (int r = 0; r < 8; ++r) {
      int m = m0 + wm * 32 + half * 16 + mbase + r;
      int n = n0 + wn * 16 + nl;
      if (m < O && n < HWc) {
        float v = acc[r];
        if (g)  v *= g[m];
        if (bb) v += bb[m];
        if (act_out == 1) v = fmaxf(v, 0.0f);
        else if (act_out == 2)
          v = fminf(fmaxf(v + 3.0f, 0.0f), 6.0f) * (1.0f / 6.0f);   // h_sigmoid
        size_t oi = ((size_t)bat * CoutTot + c_off + m) * HWc + n;
        if (addsrc) v += addsrc[oi];
        Y[oi] = v;
      }
    }
  }
}

// -------------------------------------------------------------------------
// Elementwise helpers
// -------------------------------------------------------------------------
__global__ __launch_bounds__(256)
void absdiff_kernel(const float* __restrict__ a, const float* __restrict__ b,
                    float* __restrict__ y, size_t total)
{
  size_t i = (size_t)blockIdx.x * 256 + threadIdx.x;
  if (i < total) y[i] = fabsf(a[i] - b[i]);
}

__global__ __launch_bounds__(256)
void concat_qkv_kernel(const float* __restrict__ q, const float* __restrict__ k,
                       const float* __restrict__ v, float* __restrict__ y)
{
  size_t idx = (size_t)blockIdx.x * 256 + threadIdx.x;
  size_t total = (size_t)Bn * QKVn * HWn;
  if (idx >= total) return;
  int hw = (int)(idx % HWn);
  int c  = (int)((idx / HWn) % QKVn);
  int bt = (int)(idx / ((size_t)QKVn * HWn));
  float val;
  if (c < 128)       val = q[((size_t)bt * 128 + c) * HWn + hw];
  else if (c < 256)  val = k[((size_t)bt * 128 + (c - 128)) * HWn + hw];
  else               val = v[((size_t)bt * 256 + (c - 256)) * HWn + hw];
  y[idx] = val;
}

__global__ __launch_bounds__(256)
void gate_kernel(const float* __restrict__ xp, const float* __restrict__ att,
                 const float* __restrict__ qkv, float* __restrict__ out)
{
  size_t i = (size_t)blockIdx.x * 256 + threadIdx.x;
  size_t total = (size_t)Bn * Cch * HWn;
  if (i < total) out[i] = xp[i] + att[i] * qkv[i];
}

__global__ __launch_bounds__(256)
void glu_kernel(const float* __restrict__ t, float* __restrict__ y)
{
  size_t idx = (size_t)blockIdx.x * 256 + threadIdx.x;
  size_t total = (size_t)Bn * HIDn * HWn;
  if (idx >= total) return;
  int hw = (int)(idx % HWn);
  int c  = (int)((idx / HWn) % HIDn);
  int bt = (int)(idx / ((size_t)HIDn * HWn));
  float y1 = t[((size_t)bt * FFN2 + c) * HWn + hw];
  float y2 = t[((size_t)bt * FFN2 + HIDn + c) * HWn + hw];
  float gl = 0.5f * y1 * (1.0f + erff(y1 * 0.70710678118654752f));  // exact gelu
  y[idx] = gl * y2;
}

// xx = |v1-v2| + |xr1-xr2|[i,j] + |xc1-xc2|[i,l]
__global__ __launch_bounds__(256)
void xx_kernel(const float* __restrict__ v1, const float* __restrict__ v2,
               const float* __restrict__ xr1, const float* __restrict__ xr2,
               const float* __restrict__ xc1, const float* __restrict__ xc2,
               float* __restrict__ xx)
{
  size_t idx = (size_t)blockIdx.x * 256 + threadIdx.x;
  size_t total = (size_t)Bn * DHn * HWn;
  if (idx >= total) return;
  int hw = (int)(idx % HWn);
  int c  = (int)((idx / HWn) % DHn);
  int bt = (int)(idx / ((size_t)DHn * HWn));
  int h = hw >> 6, wv = hw & 63;
  int i = h >> 2;
  int j = ((h & 3) << 2) | (wv >> 4);
  int l = wv & 15;
  size_t br = (((size_t)bt * DHn + c) * 16 + i) * 16;
  float rr = fabsf(xr1[br + j] - xr2[br + j]);
  float cc = fabsf(xc1[br + l] - xc2[br + l]);
  xx[idx] = fabsf(v1[idx] - v2[idx]) + rr + cc;
}

// -------------------------------------------------------------------------
// Axial pooling (+ linear-interp positional add for q/k).
// mode 0: shunt_w -> out[b,ch,c,h] = mean_{w4} x[b,c,h,w4*4+ch]
// mode 1: shunt_h -> out[b,ch,c,w] = mean_{h4} x[b,c,ch*16+h4,w]
// out: (B, 4, Cp, 64)
// -------------------------------------------------------------------------
__global__ __launch_bounds__(256)
void pool_kernel(const float* __restrict__ x, const float* __restrict__ pos,
                 float* __restrict__ out, int Cp, int mode)
{
  size_t idx = (size_t)blockIdx.x * 256 + threadIdx.x;
  size_t total = (size_t)Bn * 4 * Cp * 64;
  if (idx >= total) return;
  int n  = (int)(idx & 63);
  int c  = (int)((idx >> 6) % Cp);
  int ch = (int)((idx / ((size_t)64 * Cp)) & 3);
  int bt = (int)(idx / ((size_t)256 * Cp));
  const float* xb = x + ((size_t)bt * Cp + c) * HWn;
  float s = 0.0f;
  if (mode == 0) {
#pragma unroll
    for (int w4 = 0; w4 < 16; ++w4) s += xb[n * 64 + w4 * 4 + ch];
  } else {
#pragma unroll
    for (int h4 = 0; h4 < 16; ++h4) s += xb[(ch * 16 + h4) * 64 + n];
  }
  s *= (1.0f / 16.0f);
  if (pos) {
    float src = fminf(fmaxf(0.25f * n - 0.375f, 0.0f), 15.0f);
    int lo = (int)floorf(src);
    int hi = (lo + 1 < 15) ? lo + 1 : 15;
    float wt = src - (float)lo;
    const float* pp = pos + ((size_t)ch * Cp + c) * 16;
    s += pp[lo] * (1.0f - wt) + pp[hi] * wt;
  }
  out[idx] = s;
}

// -------------------------------------------------------------------------
// Axial attention logits+softmax: one block per (b,chunk,head); 64 threads.
// attn (B,4,8,64,64)
// -------------------------------------------------------------------------
__global__ __launch_bounds__(64)
void axattn_kernel(const float* __restrict__ qs, const float* __restrict__ ks,
                   float* __restrict__ attn)
{
  int bid  = blockIdx.x;              // 0..511
  int head = bid & 7;
  int chnk = (bid >> 3) & 3;
  int bt   = bid >> 5;
  __shared__ float kl[16][64];
  int t = threadIdx.x;
  for (int i = t; i < 16 * 64; i += 64) {
    int d = i >> 6, n = i & 63;
    kl[d][n] = ks[(((size_t)bt * 4 + chnk) * 128 + head * 16 + d) * 64 + n];
  }
  __syncthreads();
  int nq = t;
  float qv[16];
#pragma unroll
  for (int d = 0; d < 16; ++d)
    qv[d] = qs[(((size_t)bt * 4 + chnk) * 128 + head * 16 + d) * 64 + nq];
  float lg[64];
  float mx = -3.4e38f;
  for (int nk = 0; nk < 64; ++nk) {
    float s = 0.0f;
#pragma unroll
    for (int d = 0; d < 16; ++d) s += qv[d] * kl[d][nk];
    s *= SCALEq;
    lg[nk] = s;
    mx = fmaxf(mx, s);
  }
  float sum = 0.0f;
  for (int nk = 0; nk < 64; ++nk) { float e = __expf(lg[nk] - mx); lg[nk] = e; sum += e; }
  float inv = 1.0f / sum;
  float* ap = attn + ((((size_t)bt * 4 + chnk) * 8 + head) * 64 + nq) * 64;
  for (int nk = 0; nk < 64; ++nk) ap[nk] = lg[nk] * inv;
}

// -------------------------------------------------------------------------
// Apply attention to pooled V and write in (b, DH, 16, 16) pre-conv layout:
//   ya[b, head*32+e, chunk*4 + nq/16, nq%16] = sum_nk attn[...,nq,nk]*va[nk,e]
// -------------------------------------------------------------------------
__global__ __launch_bounds__(64)
void axapply_kernel(const float* __restrict__ attn, const float* __restrict__ vs,
                    float* __restrict__ ya)
{
  int bid  = blockIdx.x;
  int head = bid & 7;
  int chnk = (bid >> 3) & 3;
  int bt   = bid >> 5;
  __shared__ float vl[64][32];
  int t = threadIdx.x;
  for (int i = t; i < 64 * 32; i += 64) {
    int nk = i >> 5, e = i & 31;
    vl[nk][e] = vs[(((size_t)bt * 4 + chnk) * 256 + head * 32 + e) * 64 + nk];
  }
  __syncthreads();
  int nq = t;
  const float* ar = attn + ((((size_t)bt * 4 + chnk) * 8 + head) * 64 + nq) * 64;
  float a[64];
  for (int nk = 0; nk < 64; ++nk) a[nk] = ar[nk];
  int cc = chnk * 4 + (nq >> 4);
  int mm = nq & 15;
  for (int e = 0; e < 32; ++e) {
    float s = 0.0f;
    for (int nk = 0; nk < 64; ++nk) s += a[nk] * vl[nk][e];
    ya[(((size_t)bt * DHn + head * 32 + e) * 16 + cc) * 16 + mm] = s;
  }
}

// -------------------------------------------------------------------------
// host-side launch helpers
// -------------------------------------------------------------------------
inline void launch_gemm(const float* Wm, const float* X, const float* g,
                        const float* b, const float* addsrc, float* Y,
                        int O, int K, int HWc, int CoutTot, int c_off,
                        int relu_in, int act_out, hipStream_t s)
{
  dim3 grid((HWc + 63) / 64, (O + 63) / 64, Bn);
  wmma_gemm_kernel<<<grid, 256, 0, s>>>(Wm, X, g, b, addsrc, Y, O, K, HWc,
                                        CoutTot, c_off, relu_in, act_out);
}

inline dim3 g1(size_t total) { return dim3((unsigned)((total + 255) / 256)); }

} // namespace

extern "C" void kernel_launch(void* const* d_in, const int* in_sizes, int n_in,
                              void* d_out, int out_size, void* d_ws, size_t ws_size,
                              hipStream_t stream)
{
  (void)in_sizes; (void)n_in; (void)out_size; (void)ws_size;
  auto F = [&](int i) { return (const float*)d_in[i]; };

  // input indices (setup_inputs dict order, params flattened in insertion order)
  const float* x1     = F(0);
  const float* x2     = F(1);
  const float* pe_w   = F(2);   const float* pe_b  = F(3);
  const float* ln1_g  = F(4);   const float* ln1_b = F(5);
  const float* ln2_g  = F(6);   const float* ln2_b = F(7);
  const float* q1_w = F(8),  *q1_g = F(9),  *q1_b = F(10);
  const float* q2_w = F(11), *q2_g = F(12), *q2_b = F(13);
  const float* k1_w = F(14), *k1_g = F(15), *k1_b = F(16);
  const float* k2_w = F(17), *k2_g = F(18), *k2_b = F(19);
  const float* v1_w = F(20), *v1_g = F(21), *v1_b = F(22);
  const float* v2_w = F(23), *v2_g = F(24), *v2_b = F(25);
  const float* pw1_w = F(26), *pw1_g = F(27), *pw1_b = F(28);
  const float* pw2_w = F(29), *pw2_g = F(30), *pw2_b = F(31);
  const float* per1_w = F(32), *per1_g = F(33), *per1_b = F(34);
  const float* per2_w = F(35), *per2_g = F(36), *per2_b = F(37);
  const float* pec1_w = F(38), *pec1_g = F(39), *pec1_b = F(40);
  const float* pec2_w = F(41), *pec2_g = F(42), *pec2_b = F(43);
  const float* proj_w = F(44), *proj_g = F(45), *proj_b = F(46);
  const float* dw1_w = F(47), *dw1_g = F(48), *dw1_b = F(49);
  const float* dw2_w = F(50), *dw2_g = F(51), *dw2_b = F(52);
  const float* pos_rowq = F(53), *pos_rowk = F(54);
  const float* pos_colq = F(55), *pos_colk = F(56);
  const float* ffn_in_w = F(57), *ffn_dw_w = F(58), *ffn_out_w = F(59);

  // workspace layout (floats)
  constexpr size_t SZ_C   = (size_t)Bn * Cch * HWn;   // 8,388,608
  constexpr size_t SZ_DH  = (size_t)Bn * DHn * HWn;   // 16,777,216
  constexpr size_t SZ_BIG = (size_t)Bn * FFN2 * HWn;  // 44,564,480
  constexpr size_t SZ_P128 = (size_t)Bn * 4 * 128 * 64;
  constexpr size_t SZ_P256 = (size_t)Bn * 4 * 256 * 64;
  constexpr size_t SZ_ATT  = (size_t)Bn * 4 * 8 * 64 * 64;
  constexpr size_t SZ_YA   = (size_t)Bn * DHn * 256;

  float* ws = (float*)d_ws;
  size_t off = 0;
  auto alloc = [&](size_t n) { float* p = ws + off; off += n; return p; };
  float* xp1  = alloc(SZ_C);
  float* xp2  = alloc(SZ_C);
  float* lnA  = alloc(SZ_C);
  float* lnB  = alloc(SZ_C);
  float* qb   = alloc(SZ_C);
  float* kb   = alloc(SZ_C);
  float* vb1  = alloc(SZ_DH);
  float* vb2  = alloc(SZ_DH);
  float* qkv1 = alloc(SZ_C);
  float* qkv2 = alloc(SZ_C);
  float* bigA = alloc(SZ_BIG);
  float* bigB = alloc(SZ_BIG);
  float* attb = alloc(SZ_C);       // att map (B,128,HW)
  float* xxb  = alloc(SZ_DH);
  float* qs   = alloc(SZ_P128);
  float* ks   = alloc(SZ_P128);
  float* vs1  = alloc(SZ_P256);
  float* vs2  = alloc(SZ_P256);
  float* axat = alloc(SZ_ATT);
  float* ya   = alloc(SZ_YA);
  float* xr1  = alloc(SZ_YA);
  float* xr2  = alloc(SZ_YA);
  float* xc1  = alloc(SZ_YA);
  float* xc2  = alloc(SZ_YA);

  float* out1 = (float*)d_out;          // x1 result
  float* out2 = out1 + SZ_C;            // x2 result

  // ---- Phase 0: positional encoding (x + dw3x3 + bias) --------------------
  dwconv3x3_kernel<<<g1(SZ_C), 256, 0, stream>>>(x1, pe_w, nullptr, pe_b, xp1,
                                                 Cch, 1, 0, (int)SZ_C);
  dwconv3x3_kernel<<<g1(SZ_C), 256, 0, stream>>>(x2, pe_w, nullptr, pe_b, xp2,
                                                 Cch, 1, 0, (int)SZ_C);

  // ---- Phase 1: LayerNorm (ln1) -------------------------------------------
  ln_c_kernel<<<g1((size_t)Bn * HWn), 256, 0, stream>>>(xp1, ln1_g, ln1_b, lnA);
  ln_c_kernel<<<g1((size_t)Bn * HWn), 256, 0, stream>>>(xp2, ln1_g, ln1_b, lnB);

  // ---- Phase 2: q/k/v 1x1 convs (WMMA GEMMs) ------------------------------
  launch_gemm(q1_w, lnA, q1_g, q1_b, nullptr, qb,  64, 128, HWn, 128,  0, 0, 0, stream);
  launch_gemm(q2_w, lnB, q2_g, q2_b, nullptr, qb,  64, 128, HWn, 128, 64, 0, 0, stream);
  launch_gemm(k1_w, lnA, k1_g, k1_b, nullptr, kb, 128, 128, HWn, 128,  0, 0, 0, stream);
  launch_gemm(k2_w, lnB, k2_g, k2_b, nullptr, bigA,128, 128, HWn, 128,  0, 0, 0, stream);
  absdiff_kernel<<<g1(SZ_C), 256, 0, stream>>>(kb, bigA, kb, SZ_C);
  launch_gemm(v1_w, lnA, v1_g, v1_b, nullptr, vb1, 256, 128, HWn, 256, 0, 0, 0, stream);
  launch_gemm(v2_w, lnB, v2_g, v2_b, nullptr, vb2, 256, 128, HWn, 256, 0, 0, 0, stream);

  // ---- Phase 3: gated qkv paths (dw3x3 -> relu -> pw 1x1) -----------------
  constexpr size_t SZ_Q = (size_t)Bn * QKVn * HWn;
  concat_qkv_kernel<<<g1(SZ_Q), 256, 0, stream>>>(qb, kb, vb1, bigA);
  dwconv3x3_kernel<<<g1(SZ_Q), 256, 0, stream>>>(bigA, dw1_w, dw1_g, dw1_b, bigB,
                                                 QKVn, 0, 1, (int)SZ_Q);
  launch_gemm(pw1_w, bigB, pw1_g, pw1_b, nullptr, qkv1, 128, 512, HWn, 128, 0, 0, 0, stream);
  concat_qkv_kernel<<<g1(SZ_Q), 256, 0, stream>>>(qb, kb, vb2, bigA);
  dwconv3x3_kernel<<<g1(SZ_Q), 256, 0, stream>>>(bigA, dw2_w, dw2_g, dw2_b, bigB,
                                                 QKVn, 0, 1, (int)SZ_Q);
  launch_gemm(pw2_w, bigB, pw2_g, pw2_b, nullptr, qkv2, 128, 512, HWn, 128, 0, 0, 0, stream);

  // ---- Phase 4: axial attention (rows) ------------------------------------
  pool_kernel<<<g1(SZ_P128), 256, 0, stream>>>(qb, pos_rowq, qs, 128, 0);
  pool_kernel<<<g1(SZ_P128), 256, 0, stream>>>(kb, pos_rowk, ks, 128, 0);
  axattn_kernel<<<512, 64, 0, stream>>>(qs, ks, axat);
  pool_kernel<<<g1(SZ_P256), 256, 0, stream>>>(vb1, nullptr, vs1, 256, 0);
  pool_kernel<<<g1(SZ_P256), 256, 0, stream>>>(vb2, nullptr, vs2, 256, 0);
  axapply_kernel<<<512, 64, 0, stream>>>(axat, vs1, ya);
  launch_gemm(per1_w, ya, per1_g, per1_b, nullptr, xr1, 256, 256, 256, 256, 0, 1, 0, stream);
  axapply_kernel<<<512, 64, 0, stream>>>(axat, vs2, ya);
  launch_gemm(per2_w, ya, per2_g, per2_b, nullptr, xr2, 256, 256, 256, 256, 0, 1, 0, stream);

  // ---- Phase 5: axial attention (cols) ------------------------------------
  pool_kernel<<<g1(SZ_P128), 256, 0, stream>>>(qb, pos_colq, qs, 128, 1);
  pool_kernel<<<g1(SZ_P128), 256, 0, stream>>>(kb, pos_colk, ks, 128, 1);
  axattn_kernel<<<512, 64, 0, stream>>>(qs, ks, axat);
  pool_kernel<<<g1(SZ_P256), 256, 0, stream>>>(vb1, nullptr, vs1, 256, 1);
  pool_kernel<<<g1(SZ_P256), 256, 0, stream>>>(vb2, nullptr, vs2, 256, 1);
  axapply_kernel<<<512, 64, 0, stream>>>(axat, vs1, ya);
  launch_gemm(pec1_w, ya, pec1_g, pec1_b, nullptr, xc1, 256, 256, 256, 256, 0, 1, 0, stream);
  axapply_kernel<<<512, 64, 0, stream>>>(axat, vs2, ya);
  launch_gemm(pec2_w, ya, pec2_g, pec2_b, nullptr, xc2, 256, 256, 256, 256, 0, 1, 0, stream);

  // ---- Phase 6: combine, project, gate ------------------------------------
  xx_kernel<<<g1(SZ_DH), 256, 0, stream>>>(vb1, vb2, xr1, xr2, xc1, xc2, xxb);
  launch_gemm(proj_w, xxb, proj_g, proj_b, nullptr, attb, 128, 256, HWn, 128, 0, 1, 2, stream);
  gate_kernel<<<g1(SZ_C), 256, 0, stream>>>(xp1, attb, qkv1, out1);
  gate_kernel<<<g1(SZ_C), 256, 0, stream>>>(xp2, attb, qkv2, out2);

  // ---- Phase 7: FFN (gated GLU), stream 1 then stream 2 -------------------
  constexpr size_t SZ_F = (size_t)Bn * FFN2 * HWn;
  // stream 1
  ln_c_kernel<<<g1((size_t)Bn * HWn), 256, 0, stream>>>(out1, ln2_g, ln2_b, lnA);
  launch_gemm(ffn_in_w, lnA, nullptr, nullptr, nullptr, bigA, FFN2, 128, HWn, FFN2, 0, 0, 0, stream);
  dwconv3x3_kernel<<<g1(SZ_F), 256, 0, stream>>>(bigA, ffn_dw_w, nullptr, nullptr,
                                                 bigB, FFN2, 0, 0, (int)SZ_F);
  glu_kernel<<<g1((size_t)Bn * HIDn * HWn), 256, 0, stream>>>(bigB, bigA);
  launch_gemm(ffn_out_w, bigA, nullptr, nullptr, out1, out1, 128, 340, HWn, 128, 0, 0, 0, stream);
  // stream 2
  ln_c_kernel<<<g1((size_t)Bn * HWn), 256, 0, stream>>>(out2, ln2_g, ln2_b, lnA);
  launch_gemm(ffn_in_w, lnA, nullptr, nullptr, nullptr, bigA, FFN2, 128, HWn, FFN2, 0, 0, 0, stream);
  dwconv3x3_kernel<<<g1(SZ_F), 256, 0, stream>>>(bigA, ffn_dw_w, nullptr, nullptr,
                                                 bigB, FFN2, 0, 0, (int)SZ_F);
  glu_kernel<<<g1((size_t)Bn * HIDn * HWn), 256, 0, stream>>>(bigB, bigA);
  launch_gemm(ffn_out_w, bigA, nullptr, nullptr, out2, out2, 128, 340, HWn, 128, 0, 0, 0, stream);
}
